// tfdiff_WiFi_7825430413782
// MI455X (gfx1250) — matrix operand order repaired and checked by reading, not compiled, not run on hardware
//
#include <hip/hip_runtime.h>
#include <math.h>

// ---- problem dims ----
#define Bsz     8
#define Nseq    512
#define INDIM   90
#define HID     512
#define HEADS   8
#define DHD     64
#define NBLOCK  8
#define CONDDIM 128
#define EMBED   256
#define MLPH    2048
#define TOK     (Bsz*Nseq)

typedef __attribute__((ext_vector_type(16))) _Float16 v16h;
typedef __attribute__((ext_vector_type(8)))  _Float16 h8;
typedef __attribute__((ext_vector_type(4)))  _Float16 h4;
typedef __attribute__((ext_vector_type(8)))  float    v8f;
typedef __attribute__((ext_vector_type(4)))  float    f4;

// ---------------- WMMA helpers (gfx1250, wave32) ----------------
__device__ __forceinline__ v8f wmma_p(v16h a, v16h b, v8f c) {
  // D = A*B + C, f16 inputs, f32 accumulate
  return __builtin_amdgcn_wmma_f32_16x16x32_f16(false, a, false, b, (short)0, c, false, false);
}
__device__ __forceinline__ v8f zero8() {
  v8f z;
#pragma unroll
  for (int j = 0; j < 8; ++j) z[j] = 0.f;
  return z;
}

// A fragment (16x32 f16), LDS row-major [row][ld]:
// lanes 0-15: row M=l, K 0..7 then 16..23; lanes 16-31: row M=l, K 8..15 then 24..31
__device__ __forceinline__ v16h frag_a(const _Float16* t, int r0, int ld, int kb, int lane) {
  int l = lane & 15, hf = lane >> 4;
  const _Float16* p = t + (r0 + l) * ld + kb + hf * 8;
  h8 a = *(const h8*)p;
  h8 b = *(const h8*)(p + 16);
  v16h r;
#pragma unroll
  for (int j = 0; j < 8; ++j) { r[j] = a[j]; r[j + 8] = b[j]; }
  return r;
}
// B fragment (32x16 f16), LDS stores B transposed: [col][ld] with K contiguous.
// lanes 0-15: col N=l, K 0..7 then 16..23; lanes 16-31: col N=l, K 8..15 then 24..31
__device__ __forceinline__ v16h frag_b(const _Float16* t, int n0, int ld, int kb, int lane) {
  int l = lane & 15, hf = lane >> 4;
  const _Float16* p = t + (n0 + l) * ld + kb + hf * 8;
  h8 a = *(const h8*)p;
  h8 b = *(const h8*)(p + 16);
  v16h r;
#pragma unroll
  for (int j = 0; j < 8; ++j) { r[j] = a[j]; r[j + 8] = b[j]; }
  return r;
}
// A fragment loaded straight from global fp32 (used for Q in attention)
__device__ __forceinline__ v16h frag_a_gf(const float* g, int row0, int stride, int col0, int kb, int lane) {
  int l = lane & 15, hf = lane >> 4;
  const float* p = g + (size_t)(row0 + l) * stride + col0 + kb + hf * 8;
  v16h r;
#pragma unroll
  for (int j = 0; j < 8; ++j) { r[j] = (_Float16)p[j]; r[j + 8] = (_Float16)p[16 + j]; }
  return r;
}

// ---------------- generic complex GEMM ----------------
// Yr = Ar@Wr - Ai@Wi + (br - bi) ; Yi = Ar@Wi + Ai@Wr + (br + bi)
// A: [T,K] planes fp32 row-major. W: [K,N] planes fp32 row-major.
#define EP_NONE  0
#define EP_SILU  1
#define EP_ROPE  2
#define EP_GATE  3
#define EP_ADD   4
#define EP_OUTIL 5

struct GemmP {
  const float *Ar, *Ai, *Wr, *Wi, *br, *bi;
  float *Yr, *Yi;
  const float *Xr, *Xi;   // residual / add planes
  const float *Gr, *Gi;   // gate planes (adaLN), stride gStride per batch
  float *Out;             // interleaved output (EP_OUTIL)
  int T, K, N;
  int epi, seqLen, gStride;
  float lmod;             // ln(max_len)/dim for rotation epilogue
};

__global__ __launch_bounds__(256) void cgemm_kernel(GemmP P) {
  __shared__ _Float16 sAr[64 * 32], sAi[64 * 32], sWr[64 * 32], sWi[64 * 32];
  int tid = threadIdx.x;
  int lane = tid & 31, wave = tid >> 5;
  int wm = wave >> 1, wn = wave & 1;
  int m0 = blockIdx.y * 64, n0 = blockIdx.x * 64;

  v8f accR[2], accN[2], accI[2];
#pragma unroll
  for (int s = 0; s < 2; ++s) { accR[s] = zero8(); accN[s] = zero8(); accI[s] = zero8(); }

  // block-uniform: rows/cols fully in-range and K vector-aligned
  const bool mnFull = (m0 + 64 <= P.T) && (n0 + 64 <= P.N) && ((P.K & 3) == 0);
  int ksteps = (P.K + 31) >> 5;

  for (int kt = 0; kt < ksteps; ++kt) {
    int k0 = kt * 32;
    if (mnFull && (k0 + 32 <= P.K)) {
      // ---- fast path: unguarded vectorized staging ----
      // A tile 64x32: 2 float4 per thread, coalesced along K
#pragma unroll
      for (int e = 0; e < 2; ++e) {
        int flat4 = tid + e * 256;
        int r = flat4 >> 3, kq = (flat4 & 7) * 4;
        size_t off = (size_t)(m0 + r) * P.K + k0 + kq;
        f4 ar = *(const f4*)(P.Ar + off);
        f4 ai = *(const f4*)(P.Ai + off);
        h4 cr, ci;
#pragma unroll
        for (int j = 0; j < 4; ++j) { cr[j] = (_Float16)ar[j]; ci[j] = (_Float16)ai[j]; }
        *(h4*)(sAr + r * 32 + kq) = cr;
        *(h4*)(sAi + r * 32 + kq) = ci;
      }
      // W tile 32x64: 2 float4 per thread, coalesced along N; store transposed [n][k]
#pragma unroll
      for (int e = 0; e < 2; ++e) {
        int flat4 = tid + e * 256;
        int k = flat4 >> 4, nq = (flat4 & 15) * 4;
        size_t off = (size_t)(k0 + k) * P.N + n0 + nq;
        f4 wr = *(const f4*)(P.Wr + off);
        f4 wi = *(const f4*)(P.Wi + off);
#pragma unroll
        for (int j = 0; j < 4; ++j) {
          sWr[(nq + j) * 32 + k] = (_Float16)wr[j];
          sWi[(nq + j) * 32 + k] = (_Float16)wi[j];
        }
      }
      if (kt + 1 < ksteps) {
        // speculative prefetch of the next K-slab (global_prefetch_b8)
        size_t pa = (size_t)(m0 + (tid & 63)) * P.K + k0 + 32;
        size_t pw = (size_t)(k0 + 32 + (tid & 31)) * P.N + n0;
        __builtin_prefetch(P.Ar + pa, 0, 1);
        __builtin_prefetch(P.Ai + pa, 0, 1);
        __builtin_prefetch(P.Wr + pw, 0, 1);
        __builtin_prefetch(P.Wi + pw, 0, 1);
      }
    } else {
      // ---- slow path: guarded scalar staging (edge tiles only) ----
#pragma unroll
      for (int e = 0; e < 8; ++e) {
        int flat = tid + e * 256;
        int r = flat >> 5, k = flat & 31;
        int gk = k0 + k;
        float vr = 0.f, vi = 0.f;
        int gr = m0 + r;
        if (gr < P.T && gk < P.K) {
          size_t off = (size_t)gr * P.K + gk;
          vr = P.Ar[off]; vi = P.Ai[off];
        }
        sAr[r * 32 + k] = (_Float16)vr; sAi[r * 32 + k] = (_Float16)vi;
        int n = flat & 63, k2 = flat >> 6;   // coalesced along N
        int gk2 = k0 + k2, gn = n0 + n;
        float wr = 0.f, wi = 0.f;
        if (gn < P.N && gk2 < P.K) {
          size_t off = (size_t)gk2 * P.N + gn;
          wr = P.Wr[off]; wi = P.Wi[off];
        }
        sWr[n * 32 + k2] = (_Float16)wr; sWi[n * 32 + k2] = (_Float16)wi;
      }
    }
    __syncthreads();
    v16h ar = frag_a(sAr, wm * 16, 32, 0, lane);
    v16h ai = frag_a(sAi, wm * 16, 32, 0, lane);
#pragma unroll
    for (int s = 0; s < 2; ++s) {
      v16h bR = frag_b(sWr, wn * 32 + s * 16, 32, 0, lane);
      v16h bI = frag_b(sWi, wn * 32 + s * 16, 32, 0, lane);
      accR[s] = wmma_p(ar, bR, accR[s]);
      accN[s] = wmma_p(ai, bI, accN[s]);   // subtracted in epilogue (f16 WMMA lacks A-negate)
      accI[s] = wmma_p(ar, bI, accI[s]);
      accI[s] = wmma_p(ai, bR, accI[s]);
    }
    __syncthreads();
  }

  int l = lane & 15, hf = lane >> 4;
#pragma unroll
  for (int s = 0; s < 2; ++s) {
    int col = n0 + wn * 32 + s * 16 + l;
    if (col >= P.N) continue;
    float bdif = 0.f, bsum = 0.f;
    if (P.br) { float r = P.br[col], i2 = P.bi[col]; bdif = r - i2; bsum = r + i2; }
#pragma unroll
    for (int g = 0; g < 8; ++g) {
      int row = m0 + wm * 16 + g + hf * 8;
      if (row >= P.T) continue;
      float vr = accR[s][g] - accN[s][g] + bdif;
      float vi = accI[s][g] + bsum;
      if (P.epi == EP_SILU) {
        vr = vr / (1.f + __expf(-vr));
        vi = vi / (1.f + __expf(-vi));
      } else if (P.epi == EP_ROPE) {
        int n = row % P.seqLen;
        float th = (float)n * __expf(-P.lmod * (float)col);
        float cs = cosf(th), sn = sinf(th);
        float nr = vr * cs - vi * sn;
        vi = vr * sn + vi * cs;
        vr = nr;
      } else if (P.epi == EP_GATE) {
        int b = row / P.seqLen;
        size_t go = (size_t)b * P.gStride + col;
        size_t ro = (size_t)row * P.N + col;
        vr = P.Xr[ro] + P.Gr[go] * vr;
        vi = P.Xi[ro] + P.Gi[go] * vi;
      } else if (P.epi == EP_ADD) {
        size_t ro = (size_t)row * P.N + col;
        vr += P.Xr[ro]; vi += P.Xi[ro];
      }
      if (P.epi == EP_OUTIL) {
        size_t o = ((size_t)row * P.N + col) * 2;
        P.Out[o] = vr; P.Out[o + 1] = vi;
      } else {
        size_t o = (size_t)row * P.N + col;
        P.Yr[o] = vr; P.Yi[o] = vi;
      }
    }
  }
}

// ---------------- flash-style complex attention ----------------
// scores = Re(<q, conj(k)>) = qr.kr + qi.ki ; real softmax mixes vr and vi.
__global__ __launch_bounds__(128) void attn_kernel(const float* qr, const float* qi,
                                                   const float* kr, const float* ki,
                                                   const float* vr, const float* vi,
                                                   float* outR, float* outI) {
  __shared__ _Float16 sKr[32 * 64], sKi[32 * 64], sVr[64 * 32], sVi[64 * 32];
  __shared__ _Float16 sP[4][16 * 32];
  int tid = threadIdx.x, lane = tid & 31, wave = tid >> 5;
  int nq = blockIdx.x & 7;
  int bh = blockIdx.x >> 3;
  int b = bh >> 3, h = bh & 7;
  int q0 = nq * 64 + wave * 16;
  const float scale = 0.125f;  // 1/sqrt(64)

  int rowBase = b * Nseq + q0;
  v16h qrA[2], qiA[2];
#pragma unroll
  for (int ks = 0; ks < 2; ++ks) {
    qrA[ks] = frag_a_gf(qr, rowBase, HID, h * DHD, ks * 32, lane);
    qiA[ks] = frag_a_gf(qi, rowBase, HID, h * DHD, ks * 32, lane);
  }

  v8f oR[4], oI[4];
#pragma unroll
  for (int nt = 0; nt < 4; ++nt) { oR[nt] = zero8(); oI[nt] = zero8(); }
  float rowmax[8], lsum[8];
#pragma unroll
  for (int g = 0; g < 8; ++g) { rowmax[g] = -1e30f; lsum[g] = 0.f; }

  for (int kt = 0; kt < 16; ++kt) {
    __syncthreads();
    // stage K/V tiles: float4 reads coalesced along DH
#pragma unroll
    for (int e = 0; e < 4; ++e) {
      int flat4 = tid + e * 128;
      int key = flat4 >> 4, dq = (flat4 & 15) * 4;
      size_t off = (size_t)(b * Nseq + kt * 32 + key) * HID + h * DHD + dq;
      f4 a = *(const f4*)(kr + off);
      f4 c = *(const f4*)(ki + off);
      h4 ha, hc;
#pragma unroll
      for (int j = 0; j < 4; ++j) { ha[j] = (_Float16)a[j]; hc[j] = (_Float16)c[j]; }
      *(h4*)(sKr + key * 64 + dq) = ha;
      *(h4*)(sKi + key * 64 + dq) = hc;
      f4 v1 = *(const f4*)(vr + off);
      f4 v2 = *(const f4*)(vi + off);
#pragma unroll
      for (int j = 0; j < 4; ++j) {       // V stored transposed [dh][key]
        sVr[(dq + j) * 32 + key] = (_Float16)v1[j];
        sVi[(dq + j) * 32 + key] = (_Float16)v2[j];
      }
    }
    __syncthreads();

    v8f S[2];
#pragma unroll
    for (int st = 0; st < 2; ++st) {
      S[st] = zero8();
#pragma unroll
      for (int ks = 0; ks < 2; ++ks) {
        v16h bR = frag_b(sKr, st * 16, 64, ks * 32, lane);
        S[st] = wmma_p(qrA[ks], bR, S[st]);
        v16h bI = frag_b(sKi, st * 16, 64, ks * 32, lane);
        S[st] = wmma_p(qiA[ks], bI, S[st]);
      }
      S[st] *= scale;
    }

    // online softmax: rows live in VGPR index, spread across 16-lane groups
    float tmax[8];
#pragma unroll
    for (int g = 0; g < 8; ++g) tmax[g] = fmaxf(S[0][g], S[1][g]);
#pragma unroll
    for (int off = 8; off > 0; off >>= 1)
#pragma unroll
      for (int g = 0; g < 8; ++g) tmax[g] = fmaxf(tmax[g], __shfl_xor(tmax[g], off, 32));
    float alpha[8];
#pragma unroll
    for (int g = 0; g < 8; ++g) {
      float nm = fmaxf(rowmax[g], tmax[g]);
      alpha[g] = __expf(rowmax[g] - nm);
      rowmax[g] = nm;
    }
#pragma unroll
    for (int g = 0; g < 8; ++g) {
      S[0][g] = __expf(S[0][g] - rowmax[g]);
      S[1][g] = __expf(S[1][g] - rowmax[g]);
    }
    float psum[8];
#pragma unroll
    for (int g = 0; g < 8; ++g) psum[g] = S[0][g] + S[1][g];
#pragma unroll
    for (int off = 8; off > 0; off >>= 1)
#pragma unroll
      for (int g = 0; g < 8; ++g) psum[g] += __shfl_xor(psum[g], off, 32);
#pragma unroll
    for (int g = 0; g < 8; ++g) lsum[g] = lsum[g] * alpha[g] + psum[g];
#pragma unroll
    for (int nt = 0; nt < 4; ++nt)
#pragma unroll
      for (int g = 0; g < 8; ++g) { oR[nt][g] *= alpha[g]; oI[nt][g] *= alpha[g]; }

    // re-layout P (C layout -> A layout) through wave-private LDS
    int l = lane & 15, hf = lane >> 4;
    _Float16* pb = sP[wave];
#pragma unroll
    for (int st = 0; st < 2; ++st)
#pragma unroll
      for (int g = 0; g < 8; ++g)
        pb[(g + hf * 8) * 32 + st * 16 + l] = (_Float16)S[st][g];
    // wave-private buffer: only need this wave's DS stores complete (no block barrier)
    asm volatile("s_wait_dscnt 0x0" ::: "memory");

    v16h pA = frag_a(pb, 0, 32, 0, lane);
#pragma unroll
    for (int nt = 0; nt < 4; ++nt) {
      v16h vB = frag_b(sVr, nt * 16, 32, 0, lane);
      oR[nt] = wmma_p(pA, vB, oR[nt]);
      v16h vB2 = frag_b(sVi, nt * 16, 32, 0, lane);
      oI[nt] = wmma_p(pA, vB2, oI[nt]);
    }
  }

  int l = lane & 15, hf = lane >> 4;
  float inv[8];
#pragma unroll
  for (int g = 0; g < 8; ++g) inv[g] = 1.f / lsum[g];
#pragma unroll
  for (int nt = 0; nt < 4; ++nt)
#pragma unroll
    for (int g = 0; g < 8; ++g) {
      int row = q0 + g + hf * 8;
      size_t off = (size_t)(b * Nseq + row) * HID + h * DHD + nt * 16 + l;
      outR[off] = oR[nt][g] * inv[g];
      outI[off] = oI[nt][g] * inv[g];
    }
}

// ---------------- LayerNorm (feature dim, per plane) + adaLN modulate ----------------
__global__ __launch_bounds__(256) void ln_mod_kernel(const float* xr, const float* xi,
                                                     const float* mr, const float* mi,
                                                     int shOff, int scOff, int mStride,
                                                     float* outR, float* outI, int seqLen) {
  __shared__ float red[256];
  int t = blockIdx.x, tid = threadIdx.x;
  size_t base = (size_t)t * HID;
  float r0 = xr[base + tid], r1 = xr[base + tid + 256];
  float i0 = xi[base + tid], i1 = xi[base + tid + 256];

  auto bsum = [&](float v) -> float {
    red[tid] = v; __syncthreads();
    for (int s = 128; s > 0; s >>= 1) {
      if (tid < s) red[tid] += red[tid + s];
      __syncthreads();
    }
    float r = red[0]; __syncthreads();
    return r;
  };
  float mur = bsum(r0 + r1) * (1.f / HID);
  float var_r = bsum((r0 - mur) * (r0 - mur) + (r1 - mur) * (r1 - mur)) * (1.f / HID);
  float mui = bsum(i0 + i1) * (1.f / HID);
  float var_i = bsum((i0 - mui) * (i0 - mui) + (i1 - mui) * (i1 - mui)) * (1.f / HID);
  float rsr = rsqrtf(var_r + 1e-6f), rsi = rsqrtf(var_i + 1e-6f);
  int b = t / seqLen;
  size_t mb = (size_t)b * mStride;
#pragma unroll
  for (int e = 0; e < 2; ++e) {
    int d = tid + e * 256;
    float shr = mr[mb + shOff + d], scr = mr[mb + scOff + d];
    float shi = mi[mb + shOff + d], sci = mi[mb + scOff + d];
    float xnr = ((e ? r1 : r0) - mur) * rsr;
    float xni = ((e ? i1 : i0) - mui) * rsi;
    outR[base + d] = xnr * (1.f + scr) + shr;
    outI[base + d] = xni * (1.f + sci) + shi;
  }
}

// ---------------- small utility kernels ----------------
__global__ void deint_kernel(const float* src, float* dr, float* di, int n) {
  int i = blockIdx.x * blockDim.x + threadIdx.x;
  if (i < n) { dr[i] = src[2 * i]; di[i] = src[2 * i + 1]; }
}
__global__ void silu_kernel(const float* sr, const float* si, float* dr, float* di, int n) {
  int i = blockIdx.x * blockDim.x + threadIdx.x;
  if (i < n) {
    float a = sr[i], b = si[i];
    dr[i] = a / (1.f + __expf(-a));
    di[i] = b / (1.f + __expf(-b));
  }
}
__global__ void ttable_kernel(const int* t, float* er, float* ei) {
  int i = blockIdx.x * blockDim.x + threadIdx.x;
  if (i < Bsz * EMBED) {
    int b = i / EMBED, d = i % EMBED;
    float lmod = 6.90775527898214f / (float)EMBED;  // ln(1000)/256
    float th = (float)t[b] * __expf(-lmod * (float)d);
    er[i] = cosf(th);
    ei[i] = sinf(th);
  }
}

// ---------------- host side ----------------
struct Lin { const float *Wr, *Wi, *br, *bi; };

static void cgemm(hipStream_t s, GemmP p) {
  dim3 g((p.N + 63) / 64, (p.T + 63) / 64), b(256);
  cgemm_kernel<<<g, b, 0, s>>>(p);
}

static GemmP mk(const float* Ar, const float* Ai, Lin W, float* Yr, float* Yi,
                int T, int K, int N, int epi) {
  GemmP p{};
  p.Ar = Ar; p.Ai = Ai; p.Wr = W.Wr; p.Wi = W.Wi; p.br = W.br; p.bi = W.bi;
  p.Yr = Yr; p.Yi = Yi;
  p.T = T; p.K = K; p.N = N; p.epi = epi;
  p.seqLen = Nseq; p.gStride = 0; p.lmod = 0.f;
  p.Xr = nullptr; p.Xi = nullptr; p.Gr = nullptr; p.Gi = nullptr; p.Out = nullptr;
  return p;
}

extern "C" void kernel_launch(void* const* d_in, const int* in_sizes, int n_in,
                              void* d_out, int out_size, void* d_ws, size_t ws_size,
                              hipStream_t stream) {
  int idx = 0;
  const float* X = (const float*)d_in[idx++];
  const int* Tstep = (const int*)d_in[idx++];
  const float* Cin = (const float*)d_in[idx++];
  auto getLin = [&]() -> Lin {
    Lin L;
    L.Wr = (const float*)d_in[idx]; L.Wi = (const float*)d_in[idx + 1];
    L.br = (const float*)d_in[idx + 2]; L.bi = (const float*)d_in[idx + 3];
    idx += 4;
    return L;
  };
  Lin p_proj = getLin(), t1 = getLin(), t2 = getLin();
  Lin c1 = getLin(), c2 = getLin(), c3 = getLin();
  Lin blk[NBLOCK][7];  // q,k,v,o,mlp1,mlp2,ada
  for (int i = 0; i < NBLOCK; ++i)
    for (int j = 0; j < 7; ++j) blk[i][j] = getLin();
  Lin f_ada = getLin(), f_lin = getLin();

  // workspace arena (fp32 planes)
  float* ws = (float*)d_ws;
  size_t off = 0;
  auto alloc = [&](size_t n) { float* p = ws + off; off += (n + 63) & ~(size_t)63; return p; };
  float* x0r = alloc((size_t)TOK * INDIM);  float* x0i = alloc((size_t)TOK * INDIM);
  float* cr  = alloc(Bsz * CONDDIM);        float* ci  = alloc(Bsz * CONDDIM);
  float* xr  = alloc((size_t)TOK * HID);    float* xi  = alloc((size_t)TOK * HID);
  float* hr  = alloc((size_t)TOK * HID);    float* hi  = alloc((size_t)TOK * HID);
  float* qrp = alloc((size_t)TOK * HID);    float* qip = alloc((size_t)TOK * HID);
  float* krp = alloc((size_t)TOK * HID);    float* kip = alloc((size_t)TOK * HID);
  float* vrp = alloc((size_t)TOK * HID);    float* vip = alloc((size_t)TOK * HID);
  float* aor = alloc((size_t)TOK * HID);    float* aoi = alloc((size_t)TOK * HID);
  float* ur  = alloc((size_t)TOK * MLPH);   float* ui  = alloc((size_t)TOK * MLPH);
  float* ttr = alloc(Bsz * EMBED);          float* tti = alloc(Bsz * EMBED);
  float* tAr = alloc(Bsz * HID);            float* tAi = alloc(Bsz * HID);
  float* ter = alloc(Bsz * HID);            float* tei = alloc(Bsz * HID);
  float* cBr = alloc(Bsz * MLPH);           float* cBi = alloc(Bsz * MLPH);
  float* cdr = alloc(Bsz * HID);            float* cdi = alloc(Bsz * HID);
  float* scr = alloc(Bsz * HID);            float* sci = alloc(Bsz * HID);
  float* mR  = alloc(Bsz * 6 * HID);        float* mI  = alloc(Bsz * 6 * HID);

  // ---- front end ----
  {
    int n = TOK * INDIM;
    deint_kernel<<<(n + 255) / 256, 256, 0, stream>>>(X, x0r, x0i, n);
    n = Bsz * CONDDIM;
    deint_kernel<<<(n + 255) / 256, 256, 0, stream>>>(Cin, cr, ci, n);
    n = Bsz * EMBED;
    ttable_kernel<<<(n + 255) / 256, 256, 0, stream>>>(Tstep, ttr, tti);
  }
  // position embedding: clinear(p_proj,x) then complex rotation
  {
    GemmP p = mk(x0r, x0i, p_proj, xr, xi, TOK, INDIM, HID, EP_ROPE);
    p.lmod = logf((float)Nseq) / (float)HID;
    cgemm(stream, p);
  }
  // diffusion embedding
  cgemm(stream, mk(ttr, tti, t1, tAr, tAi, Bsz, EMBED, HID, EP_SILU));
  cgemm(stream, mk(tAr, tAi, t2, ter, tei, Bsz, HID, HID, EP_NONE));
  // condition embedding, fused te add
  cgemm(stream, mk(cr, ci, c1, tAr, tAi, Bsz, CONDDIM, HID, EP_SILU));
  cgemm(stream, mk(tAr, tAi, c2, cBr, cBi, Bsz, HID, MLPH, EP_SILU));
  {
    GemmP p = mk(cBr, cBi, c3, cdr, cdi, Bsz, MLPH, HID, EP_ADD);
    p.Xr = ter; p.Xi = tei;
    cgemm(stream, p);
  }
  silu_kernel<<<(Bsz * HID + 255) / 256, 256, 0, stream>>>(cdr, cdi, scr, sci, Bsz * HID);

  // ---- transformer blocks ----
  for (int i = 0; i < NBLOCK; ++i) {
    Lin Lq = blk[i][0], Lk = blk[i][1], Lv = blk[i][2], Lo = blk[i][3];
    Lin Lm1 = blk[i][4], Lm2 = blk[i][5], La = blk[i][6];
    // adaLN: m = clinear(ada, silu(c)) -> [B, 6*HID] planes
    cgemm(stream, mk(scr, sci, La, mR, mI, Bsz, HID, 6 * HID, EP_NONE));
    // h = modulate(cln(x), sh1, sc1)
    ln_mod_kernel<<<TOK, 256, 0, stream>>>(xr, xi, mR, mI, 0, HID, 6 * HID, hr, hi, Nseq);
    // q,k,v
    cgemm(stream, mk(hr, hi, Lq, qrp, qip, TOK, HID, HID, EP_NONE));
    cgemm(stream, mk(hr, hi, Lk, krp, kip, TOK, HID, HID, EP_NONE));
    cgemm(stream, mk(hr, hi, Lv, vrp, vip, TOK, HID, HID, EP_NONE));
    // attention
    attn_kernel<<<Bsz * HEADS * (Nseq / 64), 128, 0, stream>>>(qrp, qip, krp, kip, vrp, vip, aor, aoi);
    // x = x + g1 * clinear(o, attn)
    {
      GemmP p = mk(aor, aoi, Lo, xr, xi, TOK, HID, HID, EP_GATE);
      p.Xr = xr; p.Xi = xi;
      p.Gr = mR + 2 * HID; p.Gi = mI + 2 * HID; p.gStride = 6 * HID;
      cgemm(stream, p);
    }
    // h = modulate(cln(x), sh2, sc2)
    ln_mod_kernel<<<TOK, 256, 0, stream>>>(xr, xi, mR, mI, 3 * HID, 4 * HID, 6 * HID, hr, hi, Nseq);
    // mlp
    cgemm(stream, mk(hr, hi, Lm1, ur, ui, TOK, HID, MLPH, EP_SILU));
    {
      GemmP p = mk(ur, ui, Lm2, xr, xi, TOK, MLPH, HID, EP_GATE);
      p.Xr = xr; p.Xi = xi;
      p.Gr = mR + 5 * HID; p.Gi = mI + 5 * HID; p.gStride = 6 * HID;
      cgemm(stream, p);
    }
  }

  // ---- final layer ----
  cgemm(stream, mk(scr, sci, f_ada, mR, mI, Bsz, HID, 2 * HID, EP_NONE));
  ln_mod_kernel<<<TOK, 256, 0, stream>>>(xr, xi, mR, mI, 0, HID, 2 * HID, hr, hi, Nseq);
  {
    GemmP p = mk(hr, hi, f_lin, nullptr, nullptr, TOK, HID, INDIM, EP_OUTIL);
    p.Out = (float*)d_out;
    cgemm(stream, p);
  }
}
